// RelativeAttention_56315611185911
// MI455X (gfx1250) — matrix operand order repaired and checked by reading, not compiled, not run on hardware
//
#include <hip/hip_runtime.h>

typedef __attribute__((ext_vector_type(16))) _Float16 v16h;
typedef __attribute__((ext_vector_type(8)))  float    v8f;

#define S_LEN   1024
#define D_MODEL 1024
#define NHEADS  16
#define DHEAD   64

union FragH { v16h v; unsigned int u[8]; };

// A-matrix 16x32 f16 layout: lanes 0-15 hold K=0..7 (v0-3), 16..23 (v4-7);
// lanes 16-31 hold K=8..15, 24..31.  Pair p (elements 2p,2p+1) starts at:
__device__ __forceinline__ int kkA(int p, int laneHi) {
  return 2 * p + ((p & 4) ? 8 : 0) + 8 * laneHi;
}
// B-matrix 32x16 f16 layout (same k order as A): lanes 0-15 hold K=0..7, 16..23;
// lanes 16-31 hold K=8..15, 24..31.
__device__ __forceinline__ int kkB(int p, int laneHi) {
  return 2 * p + ((p & 4) ? 8 : 0) + 8 * laneHi;
}

#define WMMA_F16(a, b, c) \
  __builtin_amdgcn_wmma_f32_16x16x32_f16(false, (a), false, (b), (short)0, (c), false, false)

// ---------------------------------------------------------------------------
// Kernel 0: one-shot f32 -> f16 conversion (x and Er), removes all cvt/perm
// VALU from the GEMM/attention hot loops.  4 elements per thread.
// ---------------------------------------------------------------------------
__global__ __launch_bounds__(256) void cvt_f16_kernel(
    const float* __restrict__ src, _Float16* __restrict__ dst, int n4)
{
  int i = blockIdx.x * 256 + threadIdx.x;
  if (i < n4) {
    float4 f = ((const float4*)src)[i];
    union { _Float16 h[4]; unsigned long long u; } t;
    t.h[0] = (_Float16)f.x; t.h[1] = (_Float16)f.y;
    t.h[2] = (_Float16)f.z; t.h[3] = (_Float16)f.w;
    ((unsigned long long*)dst)[i] = t.u;
  }
}

// ---------------------------------------------------------------------------
// Kernel 1: fused QKV projection.  C = X(4096x1024) * W(1024x1024) + bias,
// written as f16 in [b, h, s, d] layout.  grid = (16 heads, 32 row-blocks, 3).
// W chunks ping-pong through LDS (one barrier per K-chunk); all 4 B-fragments
// are loaded into distinct registers before the 4 back-to-back WMMAs.
// ---------------------------------------------------------------------------
__global__ __launch_bounds__(256) void qkv_proj_kernel(
    const _Float16* __restrict__ xh,
    const float* __restrict__ Wq, const float* __restrict__ Wk, const float* __restrict__ Wv,
    const float* __restrict__ bq, const float* __restrict__ bk, const float* __restrict__ bv,
    _Float16* __restrict__ qo, _Float16* __restrict__ ko, _Float16* __restrict__ vo)
{
  const int h    = blockIdx.x;   // head index == 64-col block of W
  const int mblk = blockIdx.y;   // 128-row block of X
  const int wsel = blockIdx.z;   // 0=q 1=k 2=v
  const float* W   = (wsel == 0) ? Wq : (wsel == 1 ? Wk : Wv);
  const float* bia = (wsel == 0) ? bq : (wsel == 1 ? bk : bv);
  _Float16*    out = (wsel == 0) ? qo : (wsel == 1 ? ko : vo);

  __shared__ _Float16 WT[2][64 * 34];   // ping-pong W chunk transposed [n][kk]

  const int tid    = threadIdx.x;
  const int wave   = tid >> 5;
  const int lane   = tid & 31;
  const int laneM  = lane & 15;
  const int laneHi = lane >> 4;
  const int row0   = mblk * 128 + wave * 16;

  v8f acc[4] = {};

  // stage chunk 0
  #pragma unroll
  for (int i = 0; i < 8; ++i) {
    int idx = tid + i * 256;            // 0..2047
    int kk  = idx >> 6;                 // 0..31
    int n   = idx & 63;                 // 0..63
    WT[0][n * 34 + kk] = (_Float16)W[kk * D_MODEL + h * 64 + n];
  }
  __syncthreads();

  int pb = 0;
  for (int kc = 0; kc < D_MODEL; kc += 32, pb ^= 1) {
    // stage next chunk into the other buffer while computing this one
    if (kc + 32 < D_MODEL) {
      #pragma unroll
      for (int i = 0; i < 8; ++i) {
        int idx = tid + i * 256;
        int kk  = idx >> 6;
        int n   = idx & 63;
        WT[pb ^ 1][n * 34 + kk] = (_Float16)W[(kc + 32 + kk) * D_MODEL + h * 64 + n];
      }
    }

    // A fragment straight from global xh (pairs contiguous in k)
    FragH a;
    const _Float16* xrow = xh + (long)(row0 + laneM) * D_MODEL + kc;
    #pragma unroll
    for (int p = 0; p < 8; ++p) {
      int kk = kkA(p, laneHi);
      a.u[p] = *(const unsigned int*)&xrow[kk];
    }

    // load ALL B fragments, then issue WMMAs back-to-back (no WAR hazard nops)
    FragH bfr[4];
    #pragma unroll
    for (int jt = 0; jt < 4; ++jt) {
      int n = jt * 16 + laneM;
      #pragma unroll
      for (int p = 0; p < 8; ++p)
        bfr[jt].u[p] = *(const unsigned int*)&WT[pb][n * 34 + kkB(p, laneHi)];
    }
    #pragma unroll
    for (int jt = 0; jt < 4; ++jt)
      acc[jt] = WMMA_F16(a.v, bfr[jt].v, acc[jt]);

    __syncthreads();
  }

  // bias + store f16 into [b*H + h][s][d]
  #pragma unroll
  for (int jt = 0; jt < 4; ++jt) {
    int d = jt * 16 + laneM;
    float bval = bia[h * 64 + d];
    #pragma unroll
    for (int r = 0; r < 8; ++r) {
      int grow = row0 + r + 8 * laneHi;   // global row in [0,4096)
      int bb = grow >> 10;
      int ss = grow & 1023;
      out[((long)(bb * NHEADS + h) * S_LEN + ss) * DHEAD + d] =
          (_Float16)(acc[jt][r] + bval);
    }
  }
}

// ---------------------------------------------------------------------------
// Kernel 2: causal relative-position flash attention.
// grid = (B*H = 64, S/64 = 16), 128 threads (4 waves x 16 query rows).
// Srel[s,t] = q[s] . Er[(S-1)-(s-t)]  (skew folded analytically).
// ---------------------------------------------------------------------------
__global__ __launch_bounds__(128) void rel_attn_kernel(
    const _Float16* __restrict__ q, const _Float16* __restrict__ k,
    const _Float16* __restrict__ v, const _Float16* __restrict__ ErH,
    float* __restrict__ out)
{
  const int bh   = blockIdx.x;        // b*16 + h
  const int sB   = blockIdx.y * 64;   // query strip base
  const int tid  = threadIdx.x;
  const int wave = tid >> 5;
  const int lane = tid & 31;
  const int laneM  = lane & 15;
  const int laneHi = lane >> 4;
  const int s0 = sB + wave * 16;      // this wave's 16 query rows
  const int b  = bh >> 4;
  const int h  = bh & 15;

  const _Float16* qh = q + (long)bh * S_LEN * DHEAD;
  const _Float16* kh = k + (long)bh * S_LEN * DHEAD;
  const _Float16* vh = v + (long)bh * S_LEN * DHEAD;

  __shared__ _Float16 VT[64 * 66];        // V tile transposed [d][t]
  __shared__ float    Rbuf[4][16 * 81];   // per-wave rel tile 16x80 (padded)
  __shared__ _Float16 Pbuf[4][16 * 68];   // per-wave P staging (C->A relayout)

  // Q fragments (two K=32 chunks), direct from global f16
  FragH aq0, aq1;
  {
    const _Float16* qrow = qh + (s0 + laneM) * DHEAD;
    #pragma unroll
    for (int p = 0; p < 8; ++p) {
      int kk = kkA(p, laneHi);
      aq0.u[p] = *(const unsigned int*)&qrow[kk];
      aq1.u[p] = *(const unsigned int*)&qrow[kk + 32];
    }
  }

  v8f o[4] = {};
  float mrow[8], lrow[8];
  #pragma unroll
  for (int r = 0; r < 8; ++r) { mrow[r] = -3.0e38f; lrow[r] = 0.f; }

  for (int t0 = 0; t0 <= sB; t0 += 64) {     // causal: t0 <= sB <= s0
    // cooperatively stage V tile transposed into LDS (f16)
    #pragma unroll
    for (int i = 0; i < 32; ++i) {
      int idx = tid + i * 128;
      int tt = idx >> 6, d = idx & 63;
      VT[d * 66 + tt] = vh[(t0 + tt) * DHEAD + d];
    }
    __syncthreads();

    // ---- scores = Q K^T : load all 8 K fragments, then 8 WMMAs ----
    FragH kb[4][2];
    #pragma unroll
    for (int jt = 0; jt < 4; ++jt) {
      const _Float16* krow = kh + (t0 + jt * 16 + laneM) * DHEAD;
      #pragma unroll
      for (int p = 0; p < 8; ++p) {
        int kk = kkB(p, laneHi);
        kb[jt][0].u[p] = *(const unsigned int*)&krow[kk];
        kb[jt][1].u[p] = *(const unsigned int*)&krow[kk + 32];
      }
    }
    v8f sc[4];
    #pragma unroll
    for (int jt = 0; jt < 4; ++jt) {
      v8f z = {};
      z = WMMA_F16(aq0.v, kb[jt][0].v, z);
      z = WMMA_F16(aq1.v, kb[jt][1].v, z);
      sc[jt] = z;
    }

    // ---- rel tile: R[i,c] = q[s0+i] . Er[relBase + c], c in [0,80) ----
    const int relBase = (S_LEN - 16) - s0 + t0;
    #pragma unroll
    for (int rt = 0; rt < 5; ++rt) {
      FragH b0, b1;
      int rel = relBase + rt * 16 + laneM;
      rel = rel < 0 ? 0 : (rel > S_LEN - 1 ? S_LEN - 1 : rel);   // OOB cols are masked later
      const _Float16* erow = ErH + rel * DHEAD;
      #pragma unroll
      for (int p = 0; p < 8; ++p) {
        int kk = kkB(p, laneHi);
        b0.u[p] = *(const unsigned int*)&erow[kk];
        b1.u[p] = *(const unsigned int*)&erow[kk + 32];
      }
      v8f z = {};
      z = WMMA_F16(aq0.v, b0.v, z);
      z = WMMA_F16(aq1.v, b1.v, z);
      #pragma unroll
      for (int r = 0; r < 8; ++r)
        Rbuf[wave][(r + 8 * laneHi) * 81 + rt * 16 + laneM] = z[r];
    }
    asm volatile("s_wait_dscnt 0" ::: "memory");

    // ---- gather rel, scale, causal mask, online softmax ----
    #pragma unroll
    for (int r = 0; r < 8; ++r) {
      const int i  = r + 8 * laneHi;
      const int sg = s0 + i;
      float rmax = -3.0e38f;
      #pragma unroll
      for (int jt = 0; jt < 4; ++jt) {
        int j  = jt * 16 + laneM;
        int tg = t0 + j;
        float val = (sc[jt][r] + Rbuf[wave][i * 81 + (j - i + 15)]) * 0.125f;
        if (tg > sg) val = -3.0e38f;
        sc[jt][r] = val;
        rmax = fmaxf(rmax, val);
      }
      #pragma unroll
      for (int msk = 1; msk <= 8; msk <<= 1)
        rmax = fmaxf(rmax, __shfl_xor(rmax, msk, 32));
      float mnew  = fmaxf(mrow[r], rmax);
      float alpha = __expf(mrow[r] - mnew);
      mrow[r] = mnew;
      float rsum = 0.f;
      #pragma unroll
      for (int jt = 0; jt < 4; ++jt) {
        float p = __expf(sc[jt][r] - mnew);
        sc[jt][r] = p;
        rsum += p;
        o[jt][r] *= alpha;
      }
      #pragma unroll
      for (int msk = 1; msk <= 8; msk <<= 1)
        rsum += __shfl_xor(rsum, msk, 32);
      lrow[r] = lrow[r] * alpha + rsum;
      #pragma unroll
      for (int jt = 0; jt < 4; ++jt)
        Pbuf[wave][i * 68 + jt * 16 + laneM] = (_Float16)sc[jt][r];
    }
    asm volatile("s_wait_dscnt 0" ::: "memory");

    // ---- O += P * V : load all 8 V fragments, then 8 WMMAs ----
    FragH ap0, ap1;
    #pragma unroll
    for (int p = 0; p < 8; ++p) {
      int kk = kkA(p, laneHi);
      ap0.u[p] = *(const unsigned int*)&Pbuf[wave][laneM * 68 + kk];
      ap1.u[p] = *(const unsigned int*)&Pbuf[wave][laneM * 68 + kk + 32];
    }
    FragH vb[4][2];
    #pragma unroll
    for (int jt = 0; jt < 4; ++jt) {
      int d = jt * 16 + laneM;
      #pragma unroll
      for (int p = 0; p < 8; ++p) {
        int kk = kkB(p, laneHi);
        vb[jt][0].u[p] = *(const unsigned int*)&VT[d * 66 + kk];
        vb[jt][1].u[p] = *(const unsigned int*)&VT[d * 66 + kk + 32];
      }
    }
    #pragma unroll
    for (int jt = 0; jt < 4; ++jt) {
      o[jt] = WMMA_F16(ap0.v, vb[jt][0].v, o[jt]);
      o[jt] = WMMA_F16(ap1.v, vb[jt][1].v, o[jt]);
    }
    __syncthreads();
  }

  // ---- epilogue: normalize and store f32 output [b, s, h*64 + d] ----
  #pragma unroll
  for (int r = 0; r < 8; ++r) {
    int i  = r + 8 * laneHi;
    int sg = s0 + i;
    float inv = 1.0f / lrow[r];
    #pragma unroll
    for (int jt = 0; jt < 4; ++jt) {
      int d = jt * 16 + laneM;
      out[((long)(b * S_LEN + sg)) * D_MODEL + h * DHEAD + d] = o[jt][r] * inv;
    }
  }
}

// ---------------------------------------------------------------------------
extern "C" void kernel_launch(void* const* d_in, const int* in_sizes, int n_in,
                              void* d_out, int out_size, void* d_ws, size_t ws_size,
                              hipStream_t stream) {
  (void)in_sizes; (void)n_in; (void)out_size; (void)ws_size;
  const float* x  = (const float*)d_in[0];
  const float* Wq = (const float*)d_in[1];
  const float* bq = (const float*)d_in[2];
  const float* Wk = (const float*)d_in[3];
  const float* bk = (const float*)d_in[4];
  const float* Wv = (const float*)d_in[5];
  const float* bv = (const float*)d_in[6];
  const float* Er = (const float*)d_in[7];
  float* out = (float*)d_out;

  const size_t n = (size_t)4096 * 1024;     // B*S * D_MODEL elements per tensor
  _Float16* qws = (_Float16*)d_ws;          // 8 MB
  _Float16* kws = qws + n;                  // 8 MB
  _Float16* vws = kws + n;                  // 8 MB
  _Float16* xh  = vws + n;                  // 8 MB  (x in f16)
  _Float16* erh = xh + n;                   // 128 KB (Er in f16)

  cvt_f16_kernel<<<(int)(n / 4 / 256), 256, 0, stream>>>(x, xh, (int)(n / 4));
  cvt_f16_kernel<<<(S_LEN * DHEAD / 4 + 255) / 256, 256, 0, stream>>>(
      Er, erh, S_LEN * DHEAD / 4);
  qkv_proj_kernel<<<dim3(NHEADS, 32, 3), 256, 0, stream>>>(
      xh, Wq, Wk, Wv, bq, bk, bv, qws, kws, vws);
  rel_attn_kernel<<<dim3(4 * NHEADS, S_LEN / 64), 128, 0, stream>>>(
      qws, kws, vws, erh, out);
}
